// Sent_Posit_Drmm_Modeler_15015205666875
// MI455X (gfx1250) — compile-verified, matrix operand order checked by reading, not run on hardware
//
#include <hip/hip_runtime.h>
#include <hip/hip_bf16.h>

typedef __attribute__((ext_vector_type(16))) _Float16 v16h;
typedef __attribute__((ext_vector_type(8)))  float    v8f;

#define E_DIM    300
#define E_PAD    320
#define O_PAD    320   // padded output channels (20 tiles of 16, 5 groups of 4)
#define N_CGRP   5     // column groups of 64
#define K_CHUNKS 10    // 320 / 32
#define KPOOL    5

union FragU { v16h v; uint4 u[2]; };

// Load a 16x32 f16 fragment (A layout; B uses the same layout on B^T since
// lane = n and VGPRs hold K). Per CDNA5 ISA 7.12.2:
//   lane<16  : row = lane,    K = {k0..k0+7, k0+16..k0+23}
//   lane>=16 : row = lane-16, K = {k0+8..k0+15, k0+24..k0+31}
// i.e. two contiguous 16-byte loads per lane from row-major [rows][E_PAD] f16.
__device__ inline v16h load_frag16(const _Float16* __restrict__ base, int row0,
                                   int k0, int ld, int nvalid) {
  int lane = threadIdx.x & 31;
  int r  = row0 + (lane & 15);
  int kg = (lane >> 4) << 3;   // 0 or 8
  FragU f;
  if (r < nvalid) {
    const uint4* p = reinterpret_cast<const uint4*>(base + (size_t)r * ld + k0 + kg);
    f.u[0] = p[0];   // halves 0..7   -> K = k0+kg .. k0+kg+7
    f.u[1] = p[2];   // halves 8..15  -> K = k0+kg+16 .. k0+kg+23
  } else {
    f.u[0] = make_uint4(0u, 0u, 0u, 0u);
    f.u[1] = make_uint4(0u, 0u, 0u, 0u);
  }
  return f.v;
}

// ---------------------------------------------------------------------------
// Prepack conv weights: Wp[tap][o][e] = conv_w[o][e][tap] (f16, zero padded),
// bias_pad[o] = conv_b[o] (f32, zero padded). O padded to 320 rows.
__global__ void pack_w_kernel(const float* __restrict__ conv_w,
                              const float* __restrict__ conv_b,
                              _Float16* __restrict__ Wp,
                              float* __restrict__ bias_pad) {
  int o = blockIdx.x, tap = blockIdx.y, e = threadIdx.x;
  float v = 0.f;
  if (o < E_DIM && e < E_DIM) v = conv_w[((size_t)o * E_DIM + e) * 3 + tap];
  Wp[((size_t)tap * O_PAD + o) * E_PAD + e] = (_Float16)v;
  if (tap == 0 && e == 0) bias_pad[o] = (o < E_DIM) ? conv_b[o] : 0.f;
}

// ---------------------------------------------------------------------------
// Gather embedding rows -> f16 padded buffers; also zero the pad columns of
// the conv-output buffers (re-done every launch so workspace state never
// leaks across replays).
__global__ void gather_cast_kernel(const int* __restrict__ doc1,
                                   const int* __restrict__ question,
                                   const float* __restrict__ embeds,
                                   _Float16* __restrict__ d_h,  _Float16* __restrict__ q_h,
                                   _Float16* __restrict__ dconv, _Float16* __restrict__ qconv,
                                   int D) {
  int row = blockIdx.x, e = threadIdx.x;
  int r, idx;
  _Float16 *dst, *dstc;
  if (row < D) { r = row;      idx = doc1[r];     dst = d_h; dstc = dconv; }
  else         { r = row - D;  idx = question[r]; dst = q_h; dstc = qconv; }
  float v = (e < E_DIM) ? embeds[(size_t)idx * E_DIM + e] : 0.f;
  dst[(size_t)r * E_PAD + e] = (_Float16)v;
  if (e >= E_DIM) dstc[(size_t)r * E_PAD + e] = (_Float16)0.f;
}

// ---------------------------------------------------------------------------
// Conv1d as GEMM: y[t,o] = sigmoid( sum_{tap,k} x[t+1+tap,k] * W[o,k,tap] + b[o] ) + x[t,o]
// One wave per 16-row x 64-col output block: 4 accumulators, A fragment
// reused 4x -> 1.25 fragment loads per WMMA (120 WMMAs per wave).
__global__ __launch_bounds__(32) void conv_wmma_kernel(
    const _Float16* __restrict__ x, const _Float16* __restrict__ Wp,
    const float* __restrict__ bias, _Float16* __restrict__ y, int L) {
  int row0 = blockIdx.x * 16;
  int colg = blockIdx.y * 64;
  v8f acc[4] = {};
  #pragma unroll
  for (int tap = 0; tap < 3; ++tap) {
    const _Float16* wb = Wp + (size_t)tap * O_PAD * E_PAD;
    #pragma unroll
    for (int kc = 0; kc < K_CHUNKS; ++kc) {
      v16h a = load_frag16(x, row0 + 1 + tap, kc * 32, E_PAD, L);
      #pragma unroll
      for (int c = 0; c < 4; ++c) {
        v16h b = load_frag16(wb, colg + c * 16, kc * 32, E_PAD, O_PAD);
        acc[c] = __builtin_amdgcn_wmma_f32_16x16x32_f16(false, a, false, b,
                                                        (short)0, acc[c], false, false);
      }
    }
  }
  int lane  = threadIdx.x & 31;
  int rbase = row0 + ((lane >> 4) << 3);   // C layout: VGPR v -> M=v (+8 for hi half)
  #pragma unroll
  for (int c = 0; c < 4; ++c) {
    int col = colg + c * 16 + (lane & 15);
    float bv = bias[col];
    bool colok = (col < E_DIM);
    #pragma unroll
    for (int v = 0; v < 8; ++v) {
      int row = rbase + v;
      if (row < L && colok) {
        float z = acc[c][v] + bv;
        float s = 1.0f / (1.0f + expf(-z));
        float o = s + (float)x[(size_t)row * E_PAD + col];
        y[(size_t)row * E_PAD + col] = (_Float16)o;
      }
    }
  }
}

// ---------------------------------------------------------------------------
// Deterministic row sum-of-squares over the f16 buffers (pad cols are zero).
__global__ void row_sumsq_kernel(const _Float16* __restrict__ m,
                                 float* __restrict__ sq, int rows) {
  __shared__ float red[E_PAD];
  int row = blockIdx.x, e = threadIdx.x;
  float v = (float)m[(size_t)row * E_PAD + e];
  red[e] = v * v;
  __syncthreads();
  if (e < 64) red[e] = red[e] + red[e + 64] + red[e + 128] + red[e + 192] + red[e + 256];
  __syncthreads();
  if (e < 32) {
    float s = red[e] + red[e + 32];
    #pragma unroll
    for (int off = 16; off; off >>= 1) s += __shfl_xor(s, off, 32);
    if (e == 0) sq[row] = s;
  }
  (void)rows;
}

// ---------------------------------------------------------------------------
// Cosine similarity: sim[q,d] = (qm[q].dm[d]) * rsqrt(|qm[q]|^2) * rsqrt(|dm[d]|^2)
// One wave per 64 d-columns x both 16-row Q tiles (Q == 32): 8 accumulators,
// per k-chunk 2 A + 4 B loads feed 8 WMMAs (0.75 loads per WMMA).
__global__ __launch_bounds__(32) void sim_wmma_kernel(
    const _Float16* __restrict__ qm, const _Float16* __restrict__ dm,
    const float* __restrict__ qsq, const float* __restrict__ dsq,
    float* __restrict__ sim, int D) {
  int colg = blockIdx.x * 64;
  v8f acc[8] = {};   // acc[2*c + qtile]
  #pragma unroll
  for (int kc = 0; kc < K_CHUNKS; ++kc) {
    v16h a0 = load_frag16(qm, 0,  kc * 32, E_PAD, 32);
    v16h a1 = load_frag16(qm, 16, kc * 32, E_PAD, 32);
    #pragma unroll
    for (int c = 0; c < 4; ++c) {
      v16h b = load_frag16(dm, colg + c * 16, kc * 32, E_PAD, D);
      acc[2 * c]     = __builtin_amdgcn_wmma_f32_16x16x32_f16(false, a0, false, b,
                                                              (short)0, acc[2 * c], false, false);
      acc[2 * c + 1] = __builtin_amdgcn_wmma_f32_16x16x32_f16(false, a1, false, b,
                                                              (short)0, acc[2 * c + 1], false, false);
    }
  }
  int lane  = threadIdx.x & 31;
  int rbase = (lane >> 4) << 3;
  #pragma unroll
  for (int c = 0; c < 4; ++c) {
    int col = colg + c * 16 + (lane & 15);
    if (col < D) {
      float invd = rsqrtf(dsq[col]);
      #pragma unroll
      for (int v = 0; v < 8; ++v) {
        int r0 = rbase + v;
        sim[(size_t)r0 * D + col] = acc[2 * c][v] * rsqrtf(qsq[r0]) * invd;
        int r1 = r0 + 16;
        sim[(size_t)r1 * D + col] = acc[2 * c + 1][v] * rsqrtf(qsq[r1]) * invd;
      }
    }
  }
}

// ---------------------------------------------------------------------------
// Top-5 pool per query row: feats[q][outCol] = max, feats[q][outCol+1] = mean(top5).
__global__ void topk_kernel(const float* __restrict__ base, long rowStride,
                            long colStride, int D, float* __restrict__ feats,
                            int outCol) {
  __shared__ float red[256 * KPOOL];
  int q = blockIdx.x, t = threadIdx.x;
  const float* p = base + (size_t)q * (size_t)rowStride;
  float t5[KPOOL];
  #pragma unroll
  for (int i = 0; i < KPOOL; ++i) t5[i] = -3.4e38f;
  for (int d = t; d < D; d += 256) {
    float v = p[(size_t)d * (size_t)colStride];
    if (v > t5[KPOOL - 1]) {
      t5[KPOOL - 1] = v;
      #pragma unroll
      for (int i = KPOOL - 1; i > 0; --i)
        if (t5[i] > t5[i - 1]) { float tmp = t5[i]; t5[i] = t5[i - 1]; t5[i - 1] = tmp; }
    }
  }
  #pragma unroll
  for (int i = 0; i < KPOOL; ++i) red[t * KPOOL + i] = t5[i];
  __syncthreads();
  if (t == 0) {
    float g5[KPOOL];
    #pragma unroll
    for (int i = 0; i < KPOOL; ++i) g5[i] = -3.4e38f;
    for (int j = 0; j < 256 * KPOOL; ++j) {
      float v = red[j];
      if (v > g5[KPOOL - 1]) {
        g5[KPOOL - 1] = v;
        #pragma unroll
        for (int i = KPOOL - 1; i > 0; --i)
          if (g5[i] > g5[i - 1]) { float tmp = g5[i]; g5[i] = g5[i - 1]; g5[i - 1] = tmp; }
      }
    }
    float sum = 0.f;
    #pragma unroll
    for (int i = 0; i < KPOOL; ++i) sum += g5[i];
    feats[q * 6 + outCol]     = g5[0];
    feats[q * 6 + outCol + 1] = sum / (float)KPOOL;
  }
}

// ---------------------------------------------------------------------------
// MLP head + sigmoid-mean + BCE-with-logits. One wave (Q == 32).
__global__ void head_kernel(const float* __restrict__ feats,
                            const float* __restrict__ W1, const float* __restrict__ b1,
                            const float* __restrict__ W2, const float* __restrict__ b2,
                            const float* __restrict__ a1, const float* __restrict__ target,
                            float* __restrict__ out, int Q) {
  int q = threadIdx.x;
  float s = 0.f;
  if (q < Q) {
    float f[6];
    #pragma unroll
    for (int i = 0; i < 6; ++i) f[i] = feats[q * 6 + i];
    float alpha = a1[0];
    float z2 = b2[0];
    #pragma unroll
    for (int j = 0; j < 8; ++j) {
      float z = b1[j];
      #pragma unroll
      for (int i = 0; i < 6; ++i) z += f[i] * W1[j * 6 + i];
      z = (z > 0.f) ? z : alpha * z;     // PReLU
      z2 += z * W2[j];
    }
    s = 1.f / (1.f + expf(-z2));
  }
  #pragma unroll
  for (int off = 16; off; off >>= 1) s += __shfl_xor(s, off, 32);
  if (q == 0) {
    float emit = s / (float)Q;
    float x = emit, t = target[0];
    float lsp = -logf(1.f + expf(-x));   // log_sigmoid(x)
    float lsn = -logf(1.f + expf(x));    // log_sigmoid(-x)
    out[0] = -(t * lsp + (1.f - t) * lsn);
    out[1] = emit;
  }
}

// ---------------------------------------------------------------------------
extern "C" void kernel_launch(void* const* d_in, const int* in_sizes, int n_in,
                              void* d_out, int out_size, void* d_ws, size_t ws_size,
                              hipStream_t stream) {
  const int*   doc1     = (const int*)  d_in[0];
  const int*   question = (const int*)  d_in[1];
  const float* doc1_sim = (const float*)d_in[2];
  const float* target   = (const float*)d_in[3];
  const float* embeds   = (const float*)d_in[4];
  const float* conv_w   = (const float*)d_in[5];
  const float* conv_b   = (const float*)d_in[6];
  const float* W1       = (const float*)d_in[7];
  const float* b1       = (const float*)d_in[8];
  const float* W2       = (const float*)d_in[9];
  const float* b2       = (const float*)d_in[10];
  const float* a1       = (const float*)d_in[11];
  // d_in[12] (a2) is unused by the reference.

  const int D = in_sizes[0];   // 65536
  const int Q = in_sizes[1];   // 32

  char* ws = (char*)d_ws;
  size_t off = 0;
  auto alloc = [&](size_t bytes) -> char* {
    char* p = ws + off;
    off = (off + bytes + 255) & ~(size_t)255;
    return p;
  };
  _Float16* d_h      = (_Float16*)alloc((size_t)D * E_PAD * sizeof(_Float16));
  _Float16* dconv    = (_Float16*)alloc((size_t)D * E_PAD * sizeof(_Float16));
  _Float16* q_h      = (_Float16*)alloc((size_t)Q * E_PAD * sizeof(_Float16));
  _Float16* qconv    = (_Float16*)alloc((size_t)Q * E_PAD * sizeof(_Float16));
  _Float16* Wp       = (_Float16*)alloc((size_t)3 * O_PAD * E_PAD * sizeof(_Float16));
  float*    bias_pad = (float*)   alloc((size_t)O_PAD * sizeof(float));
  float*    dsq_in   = (float*)   alloc((size_t)D * sizeof(float));
  float*    dsq_sn   = (float*)   alloc((size_t)D * sizeof(float));
  float*    qsq_in   = (float*)   alloc((size_t)Q * sizeof(float));
  float*    qsq_sn   = (float*)   alloc((size_t)Q * sizeof(float));
  float*    sim_in   = (float*)   alloc((size_t)Q * D * sizeof(float));
  float*    sim_sn   = (float*)   alloc((size_t)Q * D * sizeof(float));
  float*    feats    = (float*)   alloc((size_t)Q * 6 * sizeof(float));
  (void)ws_size; (void)n_in; (void)out_size;

  pack_w_kernel<<<dim3(O_PAD, 3), E_PAD, 0, stream>>>(conv_w, conv_b, Wp, bias_pad);
  gather_cast_kernel<<<D + Q, E_PAD, 0, stream>>>(doc1, question, embeds,
                                                  d_h, q_h, dconv, qconv, D);
  conv_wmma_kernel<<<dim3((D + 15) / 16, N_CGRP), 32, 0, stream>>>(d_h, Wp, bias_pad, dconv, D);
  conv_wmma_kernel<<<dim3((Q + 15) / 16, N_CGRP), 32, 0, stream>>>(q_h, Wp, bias_pad, qconv, Q);
  row_sumsq_kernel<<<D, E_PAD, 0, stream>>>(d_h,   dsq_in, D);
  row_sumsq_kernel<<<D, E_PAD, 0, stream>>>(dconv, dsq_sn, D);
  row_sumsq_kernel<<<Q, E_PAD, 0, stream>>>(q_h,   qsq_in, Q);
  row_sumsq_kernel<<<Q, E_PAD, 0, stream>>>(qconv, qsq_sn, Q);
  sim_wmma_kernel<<<(D + 63) / 64, 32, 0, stream>>>(q_h,   d_h,   qsq_in, dsq_in, sim_in, D);
  sim_wmma_kernel<<<(D + 63) / 64, 32, 0, stream>>>(qconv, dconv, qsq_sn, dsq_sn, sim_sn, D);
  // sim_oh[q][d] = doc1_sim[d*Q + q]  -> rowStride 1, colStride Q
  topk_kernel<<<Q, 256, 0, stream>>>(doc1_sim, 1L,      (long)Q, D, feats, 0);
  topk_kernel<<<Q, 256, 0, stream>>>(sim_in,   (long)D, 1L,      D, feats, 2);
  topk_kernel<<<Q, 256, 0, stream>>>(sim_sn,   (long)D, 1L,      D, feats, 4);
  head_kernel<<<1, 32, 0, stream>>>(feats, W1, b1, W2, b2, a1, target, (float*)d_out, Q);
}
